// decoderUL_32504312496829
// MI455X (gfx1250) — matrix operand order repaired and checked by reading, not compiled, not run on hardware
//
#include <hip/hip_runtime.h>
#include <hip/hip_bf16.h>

// ---------------------------------------------------------------------------
// CDNA5 (gfx1250): WMMA GEMMs with LDS staging (global_load_b128 ->
// ds_store_b128 -> ds_load_b128 fragments -> v_wmma_f32_16x16x32_f16).
// Block = 256 threads = 8 waves (4x2) computing a 64x64 C macro-tile;
// each wave does 2 WMMAs per 32-wide k-chunk (1 A-frag x 2 B-frags).
// All OOB handling is branchless (clamped address + select).
// ---------------------------------------------------------------------------

typedef __attribute__((ext_vector_type(16))) _Float16 v16h;
typedef __attribute__((ext_vector_type(8)))  _Float16 v8h;
typedef __attribute__((ext_vector_type(8)))  float    v8f;

#define NG 512          // graphs
#define NN 1536         // nodes (3 per graph)
#define NE 3072         // undirected-duplicated edges
#define EHDIM 16

__device__ __forceinline__ float dev_lrelu(float v) { return v > 0.f ? v : 0.05f * v; }

__device__ __forceinline__ unsigned hash_u(unsigned x) {
    x ^= x >> 16; x *= 0x7feb352du;
    x ^= x >> 15; x *= 0x846ca68bu;
    x ^= x >> 16; return x;
}

// analytic edge list: e in [0,1536): blocks {a=(n1,n2), b=(n1,n3), c=(n2,n3)},
// e in [1536,3072): reversed copies.
__device__ __forceinline__ void edge_sd(int e, int& s, int& d) {
    int rev = (e >= NE / 2);
    int e2 = rev ? e - NE / 2 : e;
    int b = e2 / NG, g = e2 % NG;
    int n1 = 3 * g, n2 = 3 * g + 1, n3 = 3 * g + 2;
    int ss, dd;
    if (b == 0)      { ss = n1; dd = n2; }
    else if (b == 1) { ss = n1; dd = n3; }
    else             { ss = n2; dd = n3; }
    s = rev ? dd : ss;
    d = rev ? ss : dd;
}

__device__ __forceinline__ v8h pack8(const float* v) {
    v8h r;
#pragma unroll
    for (int j = 0; j < 8; ++j) r[j] = (_Float16)v[j];
    return r;
}

__device__ __forceinline__ v16h frag16(const v8h a, const v8h b) {
    v16h f;
#pragma unroll
    for (int j = 0; j < 8; ++j) { f[j] = a[j]; f[j + 8] = b[j]; }
    return f;
}

// ---------------------------------------------------------------------------
// Generic WMMA GEMM:  C[M,N] = act( Arows @ B + bias )
// A: [?,K] f32 row-major, B: [K,N] f32 row-major, bias non-null.
// mode 0: Arow(r) = A[r]
// mode 1: Arow(r) = A[3r+p0] + A[3r+p1]      (link-layer pair sum)
// mode 2: Arow(e) = A[usrc[e]] + A[udst[e]]  (symmetric edge gather-sum)
// act: 0 none, 1 leaky-relu(0.05), 2 relu.
// M must be a multiple of 64 and grid.x == M/64 (no row guards).
// ---------------------------------------------------------------------------
__global__ void __launch_bounds__(256)
wmma_gemm(const float* __restrict__ A, const float* __restrict__ B,
          const float* __restrict__ bias, float* __restrict__ C,
          int M, int N, int K, int mode, int p0, int p1, int act) {
    __shared__ __align__(16) _Float16 aL[64][32];   // [row][k]
    __shared__ __align__(16) _Float16 bL[64][32];   // transposed: [n][k]

    const int t    = threadIdx.x;        // 0..255
    const int lane = t & 31;
    const int wave = t >> 5;             // 0..7
    const int wm = wave >> 1, wn = wave & 1;   // 4x2 wave layout
    const int r16 = lane & 15, hi = lane >> 4;
    const int tm = blockIdx.x, tn = blockIdx.y;

    // staging thread roles: 64 rows x (4 threads x 8 elems) covers 64x32
    const int sRow = t >> 2;             // 0..63
    const int sCol = (t & 3) * 8;        // 0,8,16,24

    // gather rows for this staging thread's A row (always in range: M%64==0)
    const int gr = tm * 64 + sRow;
    int ra = gr, ra2 = 0;
    if (mode == 1)      { ra = 3 * gr + p0; ra2 = 3 * gr + p1; }
    else if (mode == 2) { int s, d; edge_sd(gr, s, d); ra = s; ra2 = d; }
    const int gnB = tn * 64 + sRow;      // B column staged by this thread
    const int gnC = min(gnB, N - 1);     // clamped (branchless OOB)

    v8f acc0 = {}, acc1 = {};
    for (int k0 = 0; k0 < K; k0 += 32) {
        // ---- stage A (f32 -> f16) ----
        {
            float v[8];
            if (k0 + 32 <= K) {          // uniform branch: full chunk, wide loads
                const float4 f0 = *(const float4*)&A[(long)ra * K + k0 + sCol];
                const float4 f1 = *(const float4*)&A[(long)ra * K + k0 + sCol + 4];
                v[0] = f0.x; v[1] = f0.y; v[2] = f0.z; v[3] = f0.w;
                v[4] = f1.x; v[5] = f1.y; v[6] = f1.z; v[7] = f1.w;
                if (mode != 0) {         // uniform branch
                    const float4 g0 = *(const float4*)&A[(long)ra2 * K + k0 + sCol];
                    const float4 g1 = *(const float4*)&A[(long)ra2 * K + k0 + sCol + 4];
                    v[0] += g0.x; v[1] += g0.y; v[2] += g0.z; v[3] += g0.w;
                    v[4] += g1.x; v[5] += g1.y; v[6] += g1.z; v[7] += g1.w;
                }
            } else {                     // K tail: branchless clamped scalars
#pragma unroll
                for (int j = 0; j < 8; ++j) {
                    int gk  = k0 + sCol + j;
                    int gkc = gk < K ? gk : 0;
                    float x = A[(long)ra * K + gkc];
                    if (mode != 0) x += A[(long)ra2 * K + gkc];
                    v[j] = gk < K ? x : 0.f;
                }
            }
            *(v8h*)&aL[sRow][sCol] = pack8(v);      // ds_store_b128
        }
        // ---- stage B transposed (branchless clamped scalar loads) ----
        {
            float v[8];
#pragma unroll
            for (int j = 0; j < 8; ++j) {
                int gk  = k0 + sCol + j;
                int gkc = gk < K ? gk : 0;
                float x = B[(long)gkc * N + gnC];
                v[j] = (gk < K && gnB < N) ? x : 0.f;
            }
            *(v8h*)&bL[sRow][sCol] = pack8(v);      // ds_store_b128
        }
        if (k0 + 32 < K)
            __builtin_prefetch(&B[(long)(k0 + 32) * N + gnC], 0, 1);  // global_prefetch_b8
        __syncthreads();

        // ---- fragments from LDS (ds_load_b128 pairs) + 2 WMMAs ----
        const v8h a0 = *(const v8h*)&aL[wm * 16 + r16][hi * 16];
        const v8h a1 = *(const v8h*)&aL[wm * 16 + r16][hi * 16 + 8];
        const v16h af = frag16(a0, a1);
        const v8h b00 = *(const v8h*)&bL[wn * 32 + r16][hi * 16];
        const v8h b01 = *(const v8h*)&bL[wn * 32 + r16][hi * 16 + 8];
        const v8h b10 = *(const v8h*)&bL[wn * 32 + 16 + r16][hi * 16];
        const v8h b11 = *(const v8h*)&bL[wn * 32 + 16 + r16][hi * 16 + 8];
        acc0 = __builtin_amdgcn_wmma_f32_16x16x32_f16(false, af, false, frag16(b00, b01),
                                                      (short)0, acc0, false, false);
        acc1 = __builtin_amdgcn_wmma_f32_16x16x32_f16(false, af, false, frag16(b10, b11),
                                                      (short)0, acc1, false, false);
        __syncthreads();
    }
#pragma unroll
    for (int tnn = 0; tnn < 2; ++tnn) {
        const v8f& acc = tnn ? acc1 : acc0;
        const int c = tn * 64 + wn * 32 + tnn * 16 + r16;
        const float bv = bias[min(c, N - 1)];
#pragma unroll
        for (int j = 0; j < 8; ++j) {
            int r = tm * 64 + wm * 16 + (hi ? 8 : 0) + j;
            if (c < N) {
                float v = acc[j] + bv;
                float lr = v > 0.f ? v : 0.05f * v;
                float rl = fmaxf(v, 0.f);
                v = act == 1 ? lr : (act == 2 ? rl : v);
                C[(long)r * N + c] = v;
            }
        }
    }
}

// ---------------------------------------------------------------------------
// Fused per-edge dynamic-weight message passing:
//   msg[e,h] = (Σ_k e16[e,k] * x[src(e),:] @ We2[k]  +  x[src(e),:] @ be2)[h]
//   agg[dst(e), h] += msg[e,h] * umask[e]            (global f32 atomics)
// k-scale folded into the A fragment; [E,pre,hd] never materialized.
// pre % 32 == 0, hd % 64 == 0; exact grids (no guards).
// ---------------------------------------------------------------------------
__global__ void __launch_bounds__(256)
wmma_msg(const float* __restrict__ x, const float* __restrict__ e16,
         const float* __restrict__ We2, const float* __restrict__ be2,
         const float* __restrict__ umask, float* __restrict__ agg,
         int pre, int hd) {
    __shared__ __align__(16) _Float16 aL[64][32];
    __shared__ __align__(16) _Float16 bL[64][32];   // transposed [h][p]

    const int t    = threadIdx.x;
    const int lane = t & 31;
    const int wave = t >> 5;
    const int wm = wave >> 1, wn = wave & 1;
    const int r16 = lane & 15, hi = lane >> 4;
    const int te = blockIdx.x, th = blockIdx.y;

    const int sRow = t >> 2;             // 0..63
    const int sCol = (t & 3) * 8;
    int sE, sD; edge_sd(te * 64 + sRow, sE, sD);    // staging row's src node
    const int gnB = th * 64 + sRow;                 // staged h column (< hd)

    v8f acc0 = {}, acc1 = {};
    for (int k = -1; k < EHDIM; ++k) {              // k = -1: be2 bias pass
        const float* W = (k < 0) ? be2 : (We2 + (long)k * pre * hd);
        const float scale = (k < 0) ? 1.f : e16[(long)(te * 64 + sRow) * EHDIM + k];
        for (int k0 = 0; k0 < pre; k0 += 32) {
            {   // stage A = scale * x[src] (wide loads; pre % 32 == 0)
                const float4 f0 = *(const float4*)&x[(long)sE * pre + k0 + sCol];
                const float4 f1 = *(const float4*)&x[(long)sE * pre + k0 + sCol + 4];
                float v[8] = { f0.x * scale, f0.y * scale, f0.z * scale, f0.w * scale,
                               f1.x * scale, f1.y * scale, f1.z * scale, f1.w * scale };
                *(v8h*)&aL[sRow][sCol] = pack8(v);
            }
            {   // stage B transposed (unconditional strided loads)
                float v[8];
#pragma unroll
                for (int j = 0; j < 8; ++j)
                    v[j] = W[(long)(k0 + sCol + j) * hd + gnB];
                *(v8h*)&bL[sRow][sCol] = pack8(v);
            }
            __syncthreads();
            const v8h a0 = *(const v8h*)&aL[wm * 16 + r16][hi * 16];
            const v8h a1 = *(const v8h*)&aL[wm * 16 + r16][hi * 16 + 8];
            const v16h af = frag16(a0, a1);
            const v8h b00 = *(const v8h*)&bL[wn * 32 + r16][hi * 16];
            const v8h b01 = *(const v8h*)&bL[wn * 32 + r16][hi * 16 + 8];
            const v8h b10 = *(const v8h*)&bL[wn * 32 + 16 + r16][hi * 16];
            const v8h b11 = *(const v8h*)&bL[wn * 32 + 16 + r16][hi * 16 + 8];
            acc0 = __builtin_amdgcn_wmma_f32_16x16x32_f16(false, af, false, frag16(b00, b01),
                                                          (short)0, acc0, false, false);
            acc1 = __builtin_amdgcn_wmma_f32_16x16x32_f16(false, af, false, frag16(b10, b11),
                                                          (short)0, acc1, false, false);
            __syncthreads();
        }
    }
#pragma unroll
    for (int tnn = 0; tnn < 2; ++tnn) {
        const v8f& acc = tnn ? acc1 : acc0;
        const int c = th * 64 + wn * 32 + tnn * 16 + r16;
#pragma unroll
        for (int j = 0; j < 8; ++j) {
            int er = te * 64 + wm * 16 + (hi ? 8 : 0) + j;
            int s2, d2; edge_sd(er, s2, d2);
            atomicAdd(&agg[(long)d2 * hd + c], acc[j] * umask[er]);
        }
    }
#if __has_builtin(__builtin_amdgcn_s_wait_tensorcnt)
    __builtin_amdgcn_s_wait_tensorcnt(0);
#endif
}

// ---------------------------------------------------------------------------
// Column-wise training-mode BatchNorm (biased var) + activation, in place.
// ---------------------------------------------------------------------------
__global__ void bn_act(float* __restrict__ X, int M, int N, int act) {
    __shared__ float s_sum[256], s_sq[256];
    const int col = blockIdx.x;
    float sum = 0.f, sq = 0.f;
    for (int r = threadIdx.x; r < M; r += blockDim.x) {
        float v = X[(long)r * N + col];
        sum += v; sq += v * v;
    }
    s_sum[threadIdx.x] = sum; s_sq[threadIdx.x] = sq;
    __syncthreads();
    for (int s = 128; s > 0; s >>= 1) {
        if ((int)threadIdx.x < s) {
            s_sum[threadIdx.x] += s_sum[threadIdx.x + s];
            s_sq[threadIdx.x]  += s_sq[threadIdx.x + s];
        }
        __syncthreads();
    }
    float mean = s_sum[0] / (float)M;
    float var  = s_sq[0] / (float)M - mean * mean;
    float inv  = rsqrtf(var + 1e-5f);
    for (int r = threadIdx.x; r < M; r += blockDim.x) {
        float v = (X[(long)r * N + col] - mean) * inv;
        float lr = v > 0.f ? v : 0.05f * v;
        float rl = fmaxf(v, 0.f);
        v = act == 1 ? lr : (act == 2 ? rl : v);
        X[(long)r * N + col] = v;
    }
}

// z = eps * exp(0.5*z_ls) + z_mu, deterministic hash Box-Muller
__global__ void reparam(const float* __restrict__ zmu, const float* __restrict__ zls,
                        float* __restrict__ z, int n) {
    int i = blockIdx.x * blockDim.x + threadIdx.x;
    if (i >= n) return;
    unsigned h1 = hash_u(2u * (unsigned)i + 0x9e3779b9u);
    unsigned h2 = hash_u(2u * (unsigned)i + 0x85ebca6bu);
    float u1 = ((float)(h1 >> 8) + 1.0f) * (1.0f / 16777216.0f);
    float u2 = (float)(h2 >> 8) * (1.0f / 16777216.0f);
    float eps = sqrtf(-2.0f * logf(u1)) * cosf(6.2831853f * u2);
    z[i] = eps * expf(0.5f * zls[i]) + zmu[i];
}

// per-graph softmax over 4 logits + hash categorical sample
__global__ void cat_kernel(const float* __restrict__ a1, const float* __restrict__ a2,
                           const float* __restrict__ a3, int* __restrict__ cat,
                           float* __restrict__ prob_init) {
    int g = blockIdx.x * blockDim.x + threadIdx.x;
    if (g >= NG) return;
    float e0 = (a1[2 * g + 0] + a2[2 * g + 1] + a3[2 * g + 1]) * (1.f / 3.f);
    float e1 = (a2[2 * g + 0] + a1[2 * g + 1] + a3[2 * g + 1]) * (1.f / 3.f);
    float e2 = (a3[2 * g + 0] + a2[2 * g + 1] + a1[2 * g + 1]) * (1.f / 3.f);
    float e3 = (a1[2 * g + 1] + a2[2 * g + 1] + a3[2 * g + 1]) * (1.f / 3.f);
    float m = fmaxf(fmaxf(e0, e1), fmaxf(e2, e3));
    float p[4] = { expf(e0 - m), expf(e1 - m), expf(e2 - m), expf(e3 - m) };
    float s = p[0] + p[1] + p[2] + p[3];
    float t = 0.f;
    for (int i = 0; i < 4; ++i) { p[i] = p[i] / s + 1e-4f; t += p[i]; }
    for (int i = 0; i < 4; ++i) p[i] /= t;
    float u = (float)(hash_u((unsigned)g * 2654435761u + 42u) >> 8) * (1.0f / 16777216.0f);
    int c = 3;
    float acc = 0.f;
    for (int i = 0; i < 4; ++i) { acc += p[i]; if (u < acc) { c = i; break; } }
    cat[g] = c;
    prob_init[g] = logf(p[c]);
}

__global__ void mask_kernel(const int* __restrict__ cat, float* __restrict__ umask) {
    int e = blockIdx.x * blockDim.x + threadIdx.x;
    if (e >= NE) return;
    int e2 = e % (NE / 2);
    int b = e2 / NG, g = e2 % NG;
    int c = cat[g];
    int keep = (b == 0) ? (c != 0) : (b == 1) ? (c != 2) : (c != 1);
    umask[e] = keep ? 1.f : 0.f;
}

__global__ void zero_kernel(float* __restrict__ p, int n) {
    int i = blockIdx.x * blockDim.x + threadIdx.x;
    if (i < n) p[i] = 0.f;
}

__global__ void add_kernel(float* __restrict__ dst, const float* __restrict__ src, int n) {
    int i = blockIdx.x * blockDim.x + threadIdx.x;
    if (i < n) dst[i] += src[i];
}

__global__ void concat_kernel(const float* __restrict__ fe, const float* __restrict__ x,
                              float* __restrict__ cc, int xdim) {
    int i = blockIdx.x * blockDim.x + threadIdx.x;
    int tot = 16 + xdim;
    if (i >= NN * tot) return;
    int r = i / tot, c = i % tot;
    cc[i] = (c < 16) ? fe[r * 16 + c] : x[(long)r * xdim + (c - 16)];
}

__global__ void maskmul_kernel(const float* __restrict__ raw, const float* __restrict__ umask,
                               float* __restrict__ out4) {
    int i = blockIdx.x * blockDim.x + threadIdx.x;
    if (i >= NE * 4) return;
    out4[i] = raw[i] * umask[i / 4];
}

__global__ void aux_kernel(float* __restrict__ eidx, float* __restrict__ batch,
                           float* __restrict__ ebatch) {
    int e = blockIdx.x * blockDim.x + threadIdx.x;
    if (e < NE) {
        int s, d; edge_sd(e, s, d);
        eidx[e] = (float)s;
        eidx[NE + e] = (float)d;
        ebatch[e] = (float)((e % (NE / 2)) % NG);
    }
    if (e < NN) batch[e] = (float)(e / 3);
}

// ---------------------------------------------------------------------------

extern "C" void kernel_launch(void* const* d_in, const int* in_sizes, int n_in,
                              void* d_out, int out_size, void* d_ws, size_t ws_size,
                              hipStream_t stream) {
    (void)in_sizes; (void)n_in; (void)out_size; (void)ws_size;
    const float* h    = (const float*)d_in[0];
    const float* W11  = (const float*)d_in[1];  const float* b11  = (const float*)d_in[2];
    const float* W12  = (const float*)d_in[3];  const float* b12  = (const float*)d_in[4];
    const float* Wln0 = (const float*)d_in[5];  const float* bln0 = (const float*)d_in[6];
    const float* Wl1  = (const float*)d_in[7];  const float* bl1  = (const float*)d_in[8];
    const float* Wl2  = (const float*)d_in[9];  const float* bl2  = (const float*)d_in[10];
    const float* eaW1 = (const float*)d_in[11]; const float* eab1 = (const float*)d_in[12];
    const float* eaW2 = (const float*)d_in[13]; const float* eab2 = (const float*)d_in[14];
    const float* Wf1  = (const float*)d_in[15]; const float* bf1  = (const float*)d_in[16];
    const float* Wf2  = (const float*)d_in[17]; const float* bf2  = (const float*)d_in[18];
    const float* feW1 = (const float*)d_in[19]; const float* feb1 = (const float*)d_in[20];
    const float* feW2 = (const float*)d_in[21]; const float* feb2 = (const float*)d_in[22];
    const float* Wn   = (const float*)d_in[23]; const float* bnn  = (const float*)d_in[24];
    const int convBase[3] = {25, 31, 37};
    const int preA[3] = {64, 64, 128};
    const int hdA[3]  = {64, 128, 256};

    float* out = (float*)d_out;
    float* o_node   = out + 0;        // [1536,16]
    float* o_fea    = out + 24576;    // [3072,4]
    float* o_eidx   = out + 36864;    // [2,3072]
    float* o_batch  = out + 43008;    // [1536]
    float* o_ebatch = out + 44544;    // [3072]
    float* o_prob   = out + 47616;    // [512]
    float* o_zmu    = out + 48128;    // [512,64]
    float* o_zls    = out + 80896;    // [512,64]

    float* ws = (float*)d_ws;
    float* z      = ws;                  // 32768
    float* xs     = ws + 32768;          // 98304
    float* hdl    = ws + 131072;         // 32768
    float* a1     = ws + 163840;         // 1024
    float* a2     = ws + 164864;         // 1024
    float* a3     = ws + 165888;         // 1024
    int*   cati   = (int*)(ws + 168960); // 512 ints
    float* umask  = ws + 169472;         // 3072
    float* ea     = ws + 172544;         // 49152
    float* e16    = ws + 221696;         // 49152
    float* xbuf   = ws + 270848;         // 393216
    float* xnew   = ws + 664064;         // 393216
    float* agg    = ws + 1057280;        // 393216
    float* t1     = ws + 1450496;        // 49152
    float* fe     = ws + 1499648;        // 24576
    float* cc     = ws + 1524224;        // 417792
    float* hid8   = ws + 1942016;        // 24576
    float* fraw   = ws + 1966592;        // 12288

    #define GEMM(A_, B_, bias_, C_, M_, N_, K_, mode_, p0_, p1_, act_)                         \
        wmma_gemm<<<dim3((M_) / 64, ((N_) + 63) / 64), dim3(256), 0, stream>>>(                \
            (A_), (B_), (bias_), (C_), (M_), (N_), (K_), (mode_), (p0_), (p1_), (act_))
    #define EW(kern, n_, ...) kern<<<dim3(((n_) + 255) / 256), dim3(256), 0, stream>>>(__VA_ARGS__)

    // 1) z_mu, z_ls, reparameterize
    GEMM(h, W11, b11, o_zmu, NG, 64, 128, 0, 0, 0, 0);
    GEMM(h, W12, b12, o_zls, NG, 64, 128, 0, 0, 0, 0);
    EW(reparam, NG * 64, o_zmu, o_zls, z, NG * 64);

    // 2) xs = lrelu(bn(lrelu(z @ Wln0 + b))) viewed as [1536,64]
    GEMM(z, Wln0, bln0, xs, NG, 192, 64, 0, 0, 0, 1);
    bn_act<<<dim3(64), dim3(256), 0, stream>>>(xs, NN, 64, 1);

    // 3) link scorers -> a1,a2,a3
    const int pairs[3][2] = {{0, 1}, {1, 2}, {0, 2}};
    float* aouts[3] = {a1, a2, a3};
    for (int i = 0; i < 3; ++i) {
        GEMM(xs, Wl1, bl1, hdl, NG, 64, 64, 1, pairs[i][0], pairs[i][1], 0);
        bn_act<<<dim3(64), dim3(256), 0, stream>>>(hdl, NG, 64, 1);
        GEMM(hdl, Wl2, bl2, aouts[i], NG, 2, 64, 0, 0, 0, 1);
    }

    // 4) categorical sample + edge mask
    EW(cat_kernel, NG, a1, a2, a3, cati, o_prob);
    EW(mask_kernel, NE, cati, umask);

    // 5) initial edge attributes: ea = lrelu(edge_attr(xs))
    GEMM(xs, eaW1, eab1, e16, NE, EHDIM, 64, 2, 0, 0, 0);
    bn_act<<<dim3(EHDIM), dim3(256), 0, stream>>>(e16, NE, EHDIM, 1);
    GEMM(e16, eaW2, eab2, ea, NE, EHDIM, EHDIM, 0, 0, 0, 1);

    // 6) x = xs
    hipMemcpyAsync(xbuf, xs, (size_t)NN * 64 * sizeof(float),
                   hipMemcpyDeviceToDevice, stream);

    // 7) three edge-conditioned conv layers
    float* xcur = xbuf;
    float* xnxt = xnew;
    for (int l = 0; l < 3; ++l) {
        int pre = preA[l], hd = hdA[l];
        const float* We1 = (const float*)d_in[convBase[l] + 0];
        const float* be1 = (const float*)d_in[convBase[l] + 1];
        const float* We2 = (const float*)d_in[convBase[l] + 2];
        const float* be2 = (const float*)d_in[convBase[l] + 3];
        const float* Wr  = (const float*)d_in[convBase[l] + 4];
        const float* br  = (const float*)d_in[convBase[l] + 5];

        GEMM(ea, We1, be1, e16, NE, EHDIM, EHDIM, 0, 0, 0, 0);
        bn_act<<<dim3(EHDIM), dim3(256), 0, stream>>>(e16, NE, EHDIM, 1);

        EW(zero_kernel, NN * hd, agg, NN * hd);
        wmma_msg<<<dim3(NE / 64, hd / 64), dim3(256), 0, stream>>>(
            xcur, e16, We2, be2, umask, agg, pre, hd);

        GEMM(xcur, Wr, br, xnxt, NN, hd, pre, 0, 0, 0, 0);
        EW(add_kernel, NN * hd, xnxt, agg, NN * hd);
        bn_act<<<dim3(hd), dim3(256), 0, stream>>>(xnxt, NN, hd, 2);

        float* tmp = xcur; xcur = xnxt; xnxt = tmp;
    }

    // 8) fe = lrelu(bn(lrelu(bn(x@Wf1+bf1)) @ Wf2 + bf2))
    GEMM(xcur, Wf1, bf1, t1, NN, 32, 256, 0, 0, 0, 0);
    bn_act<<<dim3(32), dim3(256), 0, stream>>>(t1, NN, 32, 1);
    GEMM(t1, Wf2, bf2, fe, NN, 16, 32, 0, 0, 0, 0);
    bn_act<<<dim3(16), dim3(256), 0, stream>>>(fe, NN, 16, 1);

    // 9) final_ea = edge_attr(concat(fe,x)) * umask
    EW(concat_kernel, NN * 272, fe, xcur, cc, 256);
    GEMM(cc, feW1, feb1, hid8, NE, 8, 272, 2, 0, 0, 0);
    bn_act<<<dim3(8), dim3(256), 0, stream>>>(hid8, NE, 8, 1);
    GEMM(hid8, feW2, feb2, fraw, NE, 4, 8, 0, 0, 0, 0);
    EW(maskmul_kernel, NE * 4, fraw, umask, o_fea);

    // 10) node_out = x @ Wn + bn
    GEMM(xcur, Wn, bnn, o_node, NN, 16, 256, 0, 0, 0, 0);

    // 11) integer-ish aux outputs (as floats)
    EW(aux_kernel, NE, o_eidx, o_batch, o_ebatch);

    #undef GEMM
    #undef EW
}
